// Transformer_32435593019961
// MI455X (gfx1250) — matrix operand order repaired and checked by reading, not compiled, not run on hardware
//
#include <hip/hip_runtime.h>
#include <cstdint>
#include <cstddef>

// ---------------------------------------------------------------------------
// Types / helpers
// ---------------------------------------------------------------------------
typedef __attribute__((ext_vector_type(16))) __bf16        v16bf;
typedef __attribute__((ext_vector_type(8)))  float         v8f;
typedef __attribute__((ext_vector_type(4)))  unsigned int  v4u;
typedef __attribute__((ext_vector_type(8)))  int           v8i;
typedef __attribute__((ext_vector_type(4)))  int           v4i;
typedef __attribute__((ext_vector_type(4)))  float         v4f;

using u16 = unsigned short;

union FragU {
  v16bf v;
  v4u   q[2];
  u16   u[16];
};

__device__ __forceinline__ u16 f2bf(float f) {
  union { float f; unsigned int u; } c{f};
  unsigned int r = c.u + 0x7FFFu + ((c.u >> 16) & 1u);   // round-to-nearest-even
  return (u16)(r >> 16);
}

__device__ __forceinline__ v16bf load_frag_pair(const u16* p0, const u16* p1) {
  FragU f;
  f.q[0] = *(const v4u*)p0;
  f.q[1] = *(const v4u*)p1;
  return f.v;
}

// A-matrix fragment (16xK slab, 16-bit): lane<16 holds K {0..15}+base,
// lane>=16 holds K {16..31}+base  (contiguous order, matches load_b_frag)
__device__ __forceinline__ v16bf load_a_frag(const u16* tile, int stride, int lane) {
  int m  = lane & 15;
  int b8 = (lane & 16) ? 16 : 0;
  const u16* r = tile + (size_t)m * stride + b8;
  return load_frag_pair(r, r + 8);
}

// B-matrix fragment (32x16, 16-bit): n = lane%16, k = (lane>=16?16:0)+j
// tile points at [n][k] layout (i.e. per-column rows of length `stride`).
__device__ __forceinline__ v16bf load_b_frag(const u16* tile, int stride, int lane) {
  int n  = lane & 15;
  int kb = (lane & 16) ? 16 : 0;
  const u16* r = tile + (size_t)n * stride + kb;
  return load_frag_pair(r, r + 8);
}

__device__ __forceinline__ v8f wmma_bf16(v16bf a, v16bf b, v8f c) {
  return __builtin_amdgcn_wmma_f32_16x16x32_bf16(false, a, false, b, (short)0, c,
                                                 false, false);
}

// Wave-local LDS ordering (cannot use __syncthreads: waves in a block run
// independent causal loop trip-counts). CDNA5 split counter: DScnt.
__device__ __forceinline__ void wave_lds_sync() {
  __builtin_amdgcn_wave_barrier();
  asm volatile("s_wait_dscnt 0x0" ::: "memory");
  __builtin_amdgcn_wave_barrier();
}

__device__ __forceinline__ unsigned int sgpr(unsigned int x) {
  return (unsigned int)__builtin_amdgcn_readfirstlane((int)x);
}

// ---------------------------------------------------------------------------
// Tensor Data Mover: async 2-D tile (rows x 32 bf16 cols) global -> LDS.
// D# per CDNA5 ISA §8.3/§8.4: group0 = {count/flags, lds_addr, gaddr[31:0],
// gaddr[56:32]|type=2}; group1 packs data_size=2B, tensor dims, tile dims and
// the 48-bit dim0 stride. Groups 2/3 unused (2-D). Tracked by TENSORcnt.
// This toolchain exposes the 6-arg builtin (extra int32x8 group before cpol).
// ---------------------------------------------------------------------------
__device__ __forceinline__ void tdm_load_tile_2d(const u16* gsrc, u16* lds_dst,
                                                 int rows, int k_stride,
                                                 int tensor_rows) {
  unsigned long long ga = (unsigned long long)(uintptr_t)gsrc;
  unsigned int lds_off  = (unsigned int)(uintptr_t)lds_dst;   // low 32 = LDS byte offset

  v4u g0;
  g0.x = sgpr(1u);                                            // count=1, user D#
  g0.y = sgpr(lds_off);                                       // lds_addr
  g0.z = sgpr((unsigned int)(ga & 0xffffffffull));            // global_addr[31:0]
  g0.w = sgpr(((unsigned int)(ga >> 32) & 0x01ffffffu) | 0x80000000u); // [56:32] | type=2

  unsigned int k  = (unsigned int)k_stride;                   // elements per row
  unsigned int td1 = (unsigned int)tensor_rows;
  v8i g1;
  g1[0] = (int)sgpr(1u << 16);                                // data_size=1 (2 bytes)
  g1[1] = (int)sgpr((k & 0xffffu) << 16);                     // tensor_dim0[15:0]
  g1[2] = (int)sgpr((k >> 16) | ((td1 & 0xffffu) << 16));     // dim0 hi | dim1 lo
  g1[3] = (int)sgpr((td1 >> 16) | (32u << 16));               // dim1 hi | tile_dim0=32
  g1[4] = (int)sgpr((unsigned int)rows);                      // tile_dim1, tile_dim2=0
  g1[5] = (int)sgpr(k);                                       // tensor_dim0_stride[31:0]
  g1[6] = (int)sgpr(0u);                                      // stride hi | dim1_stride lo
  g1[7] = (int)sgpr(0u);

  v4i gz4 = {0, 0, 0, 0};
  v8i gz8 = {0, 0, 0, 0, 0, 0, 0, 0};
  __builtin_amdgcn_tensor_load_to_lds(g0, g1, gz4, gz4, gz8, 0);
}

// ---------------------------------------------------------------------------
// fp32 -> bf16 conversion (weights)
// ---------------------------------------------------------------------------
__global__ void cvt_kernel(const float* __restrict__ in, u16* __restrict__ out, int n) {
  int i = blockIdx.x * blockDim.x + threadIdx.x;
  if (i < n) out[i] = f2bf(in[i]);
}

// ---------------------------------------------------------------------------
// LayerNorm over 1024 columns, one block (256 thr) per row.
// ---------------------------------------------------------------------------
template <bool OUTBF>
__global__ __launch_bounds__(256) void ln_kernel(const float* __restrict__ x,
                                                 const float* __restrict__ g,
                                                 const float* __restrict__ b,
                                                 void* __restrict__ out) {
  const int row = blockIdx.x;
  const int tid = threadIdx.x;
  const float* xr = x + (size_t)row * 1024;
  v4f v = ((const v4f*)xr)[tid];
  float s  = v[0] + v[1] + v[2] + v[3];
  float s2 = v[0]*v[0] + v[1]*v[1] + v[2]*v[2] + v[3]*v[3];
#pragma unroll
  for (int off = 1; off < 32; off <<= 1) {
    s  += __shfl_xor(s,  off, 32);
    s2 += __shfl_xor(s2, off, 32);
  }
  __shared__ float rs[8], rs2[8];
  if ((tid & 31) == 0) { rs[tid >> 5] = s; rs2[tid >> 5] = s2; }
  __syncthreads();
  float S = 0.f, S2 = 0.f;
#pragma unroll
  for (int i = 0; i < 8; ++i) { S += rs[i]; S2 += rs2[i]; }
  const float mean = S * (1.0f / 1024.0f);
  const float var  = S2 * (1.0f / 1024.0f) - mean * mean;
  const float inv  = rsqrtf(var + 1e-5f);
#pragma unroll
  for (int c = 0; c < 4; ++c) {
    int col  = tid * 4 + c;
    float y  = (v[c] - mean) * inv * g[col] + b[col];
    if constexpr (OUTBF) ((u16*)out)[(size_t)row * 1024 + col] = f2bf(y);
    else                 ((float*)out)[(size_t)row * 1024 + col] = y;
  }
}

// ---------------------------------------------------------------------------
// bf16 GEMM:  C[M,N] = A[M,K] * B[K,N]   (row-major, f32 accumulate)
// Block tile 128x64, BK=32. 8 waves; wave w -> rows (w%4)*32, cols (w/4)*32,
// i.e. a 32x32 output per wave = 4 WMMAs/K-step with A/B fragment reuse.
// A tile staged by the Tensor Data Mover (wave 0, TENSORcnt); B tile staged
// cooperatively transposed for contiguous b-fragment ds_load_b128s.
// Epilogue: optional bias / exact-erf GELU / fp32 residual / bf16 or f32 out.
// ---------------------------------------------------------------------------
template <bool BIAS, bool GELU, bool RES, bool OUTBF>
__global__ __launch_bounds__(256) void gemm_kernel(
    const u16* __restrict__ A, const u16* __restrict__ B,
    const float* __restrict__ bias, const float* __restrict__ res,
    float* __restrict__ outF, u16* __restrict__ outB,
    int M, int N, int K) {
  __shared__ u16 As[128 * 32];  // [row][k]
  __shared__ u16 Bt[64 * 32];   // [col][k]  (transposed tile of B)

  const int tid  = threadIdx.x;
  const int lane = tid & 31;
  const int w    = tid >> 5;
  const int wr   = w & 3;       // 0..3 : 32-row slab
  const int wc   = w >> 2;      // 0..1 : 32-col slab
  const int brow = blockIdx.y * 128;
  const int bcol = blockIdx.x * 64;

  v8f acc00 = {}, acc01 = {}, acc10 = {}, acc11 = {};

  for (int k0 = 0; k0 < K; k0 += 32) {
    // A tile 128x32 via TDM (async DMA into LDS), issued by wave 0 only.
    if (w == 0) {
      tdm_load_tile_2d(A + (size_t)brow * K + k0, As, 128, K, M - brow);
      __builtin_amdgcn_s_wait_tensorcnt(0);
    }
    // B tile transposed (32x64 -> [col][k]), all 256 threads.
#pragma unroll
    for (int i = 0; i < 4; ++i) {
      int e  = tid + i * 256;
      int kr = e >> 5;           // 0..31
      int cu = e & 31;           // column pair
      unsigned int d =
          *(const unsigned int*)(B + (size_t)(k0 + kr) * N + bcol + cu * 2);
      Bt[(cu * 2 + 0) * 32 + kr] = (u16)(d & 0xffffu);
      Bt[(cu * 2 + 1) * 32 + kr] = (u16)(d >> 16);
      if (k0 + 32 < K)           // prefetch next B tile -> global_prefetch_b8
        __builtin_prefetch(B + (size_t)(k0 + 32 + kr) * N + bcol + cu * 2, 0, 0);
    }
    __syncthreads();

    v16bf a0 = load_a_frag(As + (wr * 32 + 0)  * 32, 32, lane);
    v16bf a1 = load_a_frag(As + (wr * 32 + 16) * 32, 32, lane);
    v16bf b0 = load_b_frag(Bt + (wc * 32 + 0)  * 32, 32, lane);
    v16bf b1 = load_b_frag(Bt + (wc * 32 + 16) * 32, 32, lane);
    acc00 = wmma_bf16(a0, b0, acc00);
    acc01 = wmma_bf16(a0, b1, acc01);
    acc10 = wmma_bf16(a1, b0, acc10);
    acc11 = wmma_bf16(a1, b1, acc11);
    __syncthreads();
  }

  // epilogue: D layout -> row = r + (lane>=16 ? 8 : 0), col = lane%16
  const int off8  = (lane & 16) ? 8 : 0;
  const int ncol0 = bcol + wc * 32 + (lane & 15);
#pragma unroll
  for (int i = 0; i < 2; ++i) {
    const int mrow = brow + wr * 32 + i * 16 + off8;
#pragma unroll
    for (int r = 0; r < 8; ++r) {
      int row = mrow + r;
#pragma unroll
      for (int t = 0; t < 2; ++t) {
        float vv;
        if (i == 0) vv = (t ? acc01[r] : acc00[r]);
        else        vv = (t ? acc11[r] : acc10[r]);
        int col = ncol0 + t * 16;
        if constexpr (BIAS) vv += bias[col];
        if constexpr (GELU) vv = 0.5f * vv * (1.0f + erff(vv * 0.70710678118f));
        if constexpr (RES)  vv += res[(size_t)row * N + col];
        if constexpr (OUTBF) outB[(size_t)row * N + col] = f2bf(vv);
        else                 outF[(size_t)row * N + col] = vv;
      }
    }
  }
}

// ---------------------------------------------------------------------------
// Flash-style causal ALiBi attention.
// qkv: [2048, 3072] bf16 (q|k|v thirds, head h at cols h*64..h*64+63)
// out: [2048, 1024] bf16.
// Block = 128 thr = 4 waves; wave w handles q-tile (blockIdx.x*4 + w) of
// head blockIdx.y. Key blocks of 32; S via 4 WMMAs, PV via 4 WMMAs.
// ---------------------------------------------------------------------------
__global__ __launch_bounds__(128) void attn_kernel(const u16* __restrict__ qkv,
                                                   u16* __restrict__ out) {
  __shared__ u16 smem[4 * 4608];          // per-wave: K 2048 | V 2048 | P 512
  const int tid  = threadIdx.x;
  const int lane = tid & 31;
  const int w    = tid >> 5;
  u16* ldsK = smem + w * 4608;
  u16* ldsV = ldsK + 2048;
  u16* ldsP = ldsK + 4096;

  const int qt   = blockIdx.x * 4 + w;    // 0..127
  const int head = blockIdx.y;            // 0..15
  const int RQ   = 3072;
  const float scale = 0.125f;                           // 64^-0.5
  const float slope = exp2f(-0.5f * (float)(head + 1)); // ALiBi, 16 heads

  // Q fragments (16x64, two K=32 slabs), straight from global (layout-aligned)
  const u16* qbase = qkv + (size_t)(qt * 16) * RQ + head * 64;
  v16bf aq0 = load_a_frag(qbase +  0, RQ, lane);
  v16bf aq1 = load_a_frag(qbase + 32, RQ, lane);

  float mI[8], lI[8];
#pragma unroll
  for (int r = 0; r < 8; ++r) { mI[r] = -1e30f; lI[r] = 0.f; }
  v8f o[4] = {v8f{}, v8f{}, v8f{}, v8f{}};

  const int off8  = (lane & 16) ? 8 : 0;
  const int n     = lane & 15;
  const int qrow0 = qt * 16 + off8;
  const int ktmax = (qt * 16 + 15) >> 5;

  for (int kt = 0; kt <= ktmax; ++kt) {
    // stage K, V blocks (32 keys x 64 feats) — one row per lane, 16B vectors
    {
      const int key = kt * 32 + lane;
      const v4u* gK = (const v4u*)(qkv + (size_t)key * RQ + 1024 + head * 64);
      const v4u* gV = (const v4u*)(qkv + (size_t)key * RQ + 2048 + head * 64);
      v4u* lK = (v4u*)(ldsK + lane * 64);
      v4u* lV = (v4u*)(ldsV + lane * 64);
#pragma unroll
      for (int i = 0; i < 8; ++i) { lK[i] = gK[i]; lV[i] = gV[i]; }
    }
    wave_lds_sync();

    // S = Q * K^T : two 16x16 score tiles (keys 0..15 and 16..31)
    v8f s0 = {};
    v8f s1 = {};
    s0 = wmma_bf16(aq0, load_b_frag(ldsK,            64, lane), s0);
    s0 = wmma_bf16(aq1, load_b_frag(ldsK + 32,       64, lane), s0);
    s1 = wmma_bf16(aq0, load_b_frag(ldsK + 16 * 64,      64, lane), s1);
    s1 = wmma_bf16(aq1, load_b_frag(ldsK + 16 * 64 + 32, 64, lane), s1);

    // scale + ALiBi + causal mask + online softmax (fp32)
    const int key0 = kt * 32 + n;
    const int key1 = key0 + 16;
    float p0[8], p1[8], alpha[8];
#pragma unroll
    for (int r = 0; r < 8; ++r) {
      const int qg = qrow0 + r;
      float v0 = s0[r] * scale + slope * (float)key0;
      float v1 = s1[r] * scale + slope * (float)key1;
      if (key0 > qg) v0 = -1e30f;
      if (key1 > qg) v1 = -1e30f;
      float tmax = fmaxf(v0, v1);
#pragma unroll
      for (int off = 1; off < 16; off <<= 1)
        tmax = fmaxf(tmax, __shfl_xor(tmax, off, 32));
      const float mnew = fmaxf(mI[r], tmax);
      v0 = __expf(v0 - mnew);
      v1 = __expf(v1 - mnew);
      const float a = __expf(mI[r] - mnew);
      mI[r] = mnew;
      float ls = v0 + v1;
#pragma unroll
      for (int off = 1; off < 16; off <<= 1) ls += __shfl_xor(ls, off, 32);
      lI[r] = lI[r] * a + ls;
      alpha[r] = a;
      p0[r] = v0;
      p1[r] = v1;
    }
#pragma unroll
    for (int r = 0; r < 8; ++r) {
      o[0][r] *= alpha[r]; o[1][r] *= alpha[r];
      o[2][r] *= alpha[r]; o[3][r] *= alpha[r];
    }

    // P (D layout) -> LDS -> A fragment for the PV product
#pragma unroll
    for (int r = 0; r < 8; ++r) {
      const int m = off8 + r;
      ldsP[m * 32 + n]      = f2bf(p0[r]);
      ldsP[m * 32 + 16 + n] = f2bf(p1[r]);
    }
    wave_lds_sync();
    v16bf ap = load_a_frag(ldsP, 32, lane);

    // O += P(16x32) * V(32x64) — four 16-col output tiles
    const int kb = (lane & 16) ? 16 : 0;
#pragma unroll
    for (int vt = 0; vt < 4; ++vt) {
      FragU bv;
      const int coln = vt * 16 + n;
#pragma unroll
      for (int j = 0; j < 16; ++j) bv.u[j] = ldsV[(kb + j) * 64 + coln];
      o[vt] = wmma_bf16(ap, bv.v, o[vt]);
    }
  }

  // normalize + store  (out col = head*64 + vt*16 + n)
#pragma unroll
  for (int vt = 0; vt < 4; ++vt) {
#pragma unroll
    for (int r = 0; r < 8; ++r) {
      const int row = qt * 16 + off8 + r;
      const int col = head * 64 + vt * 16 + n;
      out[(size_t)row * 1024 + col] = f2bf(o[vt][r] / lI[r]);
    }
  }
}

// ---------------------------------------------------------------------------
// Host driver
// ---------------------------------------------------------------------------
extern "C" void kernel_launch(void* const* d_in, const int* in_sizes, int n_in,
                              void* d_out, int out_size, void* d_ws, size_t ws_size,
                              hipStream_t stream) {
  (void)in_sizes; (void)n_in; (void)out_size; (void)ws_size;
  constexpr int NT = 2048, DM = 1024, NQ = 3072, FFD = 4096, LAY = 4, HD = 16;

  const float* x      = (const float*)d_in[0];
  const float* qkv_w  = (const float*)d_in[1];
  const float* out_w  = (const float*)d_in[2];
  const float* attn_g = (const float*)d_in[3];
  const float* attn_b = (const float*)d_in[4];
  const float* ffg    = (const float*)d_in[5];
  const float* ffb    = (const float*)d_in[6];
  const float* ff_w1  = (const float*)d_in[7];
  const float* ff_b1  = (const float*)d_in[8];
  const float* ff_w2  = (const float*)d_in[9];
  const float* ff_b2  = (const float*)d_in[10];
  const float* fin_g  = (const float*)d_in[11];
  const float* fin_b  = (const float*)d_in[12];

  char* p = (char*)d_ws;
  float* h     = (float*)p; p += (size_t)NT * DM * 4;
  u16*  ybf    = (u16*)p;   p += (size_t)NT * DM * 2;
  u16*  qkvbf  = (u16*)p;   p += (size_t)NT * NQ * 2;
  u16*  attnbf = (u16*)p;   p += (size_t)NT * DM * 2;
  u16*  ffbf   = (u16*)p;   p += (size_t)NT * FFD * 2;
  u16*  wqkv   = (u16*)p;   p += (size_t)DM * NQ * 2;
  u16*  wo     = (u16*)p;   p += (size_t)DM * DM * 2;
  u16*  w1b    = (u16*)p;   p += (size_t)DM * FFD * 2;
  u16*  w2b    = (u16*)p;   p += (size_t)FFD * DM * 2;

  (void)hipMemcpyAsync(h, x, (size_t)NT * DM * 4, hipMemcpyDeviceToDevice, stream);

  for (int l = 0; l < LAY; ++l) {
    { int nn = DM * NQ;  cvt_kernel<<<(nn + 255) / 256, 256, 0, stream>>>(qkv_w + (size_t)l * nn, wqkv, nn); }
    { int nn = DM * DM;  cvt_kernel<<<(nn + 255) / 256, 256, 0, stream>>>(out_w + (size_t)l * nn, wo,   nn); }
    { int nn = DM * FFD; cvt_kernel<<<(nn + 255) / 256, 256, 0, stream>>>(ff_w1 + (size_t)l * nn, w1b,  nn); }
    { int nn = FFD * DM; cvt_kernel<<<(nn + 255) / 256, 256, 0, stream>>>(ff_w2 + (size_t)l * nn, w2b,  nn); }

    ln_kernel<true><<<NT, 256, 0, stream>>>(h, attn_g + (size_t)l * DM, attn_b + (size_t)l * DM, ybf);

    gemm_kernel<false, false, false, true>
        <<<dim3(NQ / 64, NT / 128), 256, 0, stream>>>(ybf, wqkv, nullptr, nullptr,
                                                      nullptr, qkvbf, NT, NQ, DM);

    attn_kernel<<<dim3(NT / 64, HD), 128, 0, stream>>>(qkvbf, attnbf);

    gemm_kernel<false, false, true, false>
        <<<dim3(DM / 64, NT / 128), 256, 0, stream>>>(attnbf, wo, nullptr, h,
                                                      h, nullptr, NT, DM, DM);

    ln_kernel<true><<<NT, 256, 0, stream>>>(h, ffg + (size_t)l * DM, ffb + (size_t)l * DM, ybf);

    gemm_kernel<true, true, false, true>
        <<<dim3(FFD / 64, NT / 128), 256, 0, stream>>>(ybf, w1b, ff_b1 + (size_t)l * FFD,
                                                       nullptr, nullptr, ffbf, NT, FFD, DM);

    gemm_kernel<true, false, true, false>
        <<<dim3(DM / 64, NT / 128), 256, 0, stream>>>(ffbf, w2b, ff_b2 + (size_t)l * DM,
                                                      h, h, nullptr, NT, DM, FFD);
  }

  ln_kernel<false><<<NT, 256, 0, stream>>>(h, fin_g, fin_b, d_out);
}
